// STPAN_66529043415395
// MI455X (gfx1250) — matrix operand order repaired and checked by reading, not compiled, not run on hardware
//
#include <hip/hip_runtime.h>
#include <hip/hip_bf16.h>
#include <stdint.h>

// ---------------------------------------------------------------------------
// Shapes
// ---------------------------------------------------------------------------
#define BB 16
#define TTD 12
#define NN 307
#define HH 64
#define NHEAD 4
#define MMEM 20
#define NPB 2
#define NOUT 12
#define ROWS_BN  (BB * NN)        // 4912  (= 16 * 307, multiple of 16)
#define ROWS_BTN (BB * TTD * NN)  // 58944 (multiple of 16)

typedef __attribute__((ext_vector_type(16))) _Float16 v16h;
typedef __attribute__((ext_vector_type(8)))  float    v8f;

// ---------------------------------------------------------------------------
// Input index map.  Assumption: harness order = x_data, time_id, day_id, then
// params flattened jax-style (dict keys sorted alphabetically at each level).
// ---------------------------------------------------------------------------
enum {
  IN_XDATA = 0, IN_TIME, IN_DAY,
  E_C1B, E_C1W, E_C2B, E_C2W, E_DAYEMB, E_DENSEB, E_DENSEW,
  E_L1B, E_L1W, E_NODE, E_POS, E_TIMEEMB,            // emb: 3..14
  G_D1B, G_D1W, G_KB, G_KW,                          // gma: 15..18
  GM_C1B, GM_C1W, GM_C2B, GM_C2W, GM_M1,             // gma.mem: 19..23
  G_QB, G_QW,                                        // 24,25
  GR_C1B, GR_C1W, GR_C2B, GR_C2W, GR_M1,             // gma.resmem: 26..30
  G_STRU, G_VB, G_VW,                                // 31..33
  M_C1B, M_C1W, M_C2B, M_C2W, M_M1,                  // mem: 34..38
  P_C1B, P_C1W, P_C2B, P_C2W,                        // period: 39..42
  PRED_B, PRED_W,                                    // 43,44
  R_C1B, R_C1W, R_C2B, R_C2W                         // residual: 45..48
};

// ---------------------------------------------------------------------------
// WMMA helpers (wave32, 16x16x32 f16 -> f32 accumulate)
// A fragment (16x32, MxK): lane m = lane&15; kbase = (lane&16)?8:0;
//   elements 0..7  -> K = kbase+e      (contiguous 8 halves)
//   elements 8..15 -> K = kbase+16+e-8 (contiguous 8 halves)
// B fragment (32x16, KxN) from an [N x K]-indexed base (i.e. B[k][n]=base[n*ld+k]):
//   lane n = lane&15; kbase = (lane&16)?8:0; same K order as the A fragment
// C/D: lane col = lane&15; vgpr r -> row = ((lane&16)?8:0)+r
// ---------------------------------------------------------------------------
template <typename PtrT>
__device__ __forceinline__ v16h frag_a(PtrT base, int row, int ld, int kk, int lane) {
  int kb = (lane & 16) ? 8 : 0;
  PtrT p = base + (size_t)row * ld + kk + kb;
  v16h f;
#pragma unroll
  for (int i = 0; i < 8; ++i) { f[i] = p[i]; f[i + 8] = p[i + 16]; }
  return f;
}

template <typename PtrT>
__device__ __forceinline__ v16h frag_b(PtrT base, int col, int ld, int kk, int lane) {
  int kb = (lane & 16) ? 8 : 0;
  PtrT p = base + (size_t)col * ld + kk + kb;
  v16h f;
#pragma unroll
  for (int i = 0; i < 8; ++i) { f[i] = p[i]; f[i + 8] = p[i + 16]; }
  return f;
}

__device__ __forceinline__ v8f wmma_f16(v16h a, v16h b, v8f c) {
  return __builtin_amdgcn_wmma_f32_16x16x32_f16(false, a, false, b, (short)0, c,
                                                false, false);
}

// ---------------------------------------------------------------------------
// f32 -> f16 conversion (weights / params)
// ---------------------------------------------------------------------------
__global__ void k_f32_to_f16(const float* __restrict__ src,
                             _Float16* __restrict__ dst, int n) {
  int i = blockIdx.x * blockDim.x + threadIdx.x;
  if (i < n) dst[i] = (_Float16)src[i];
}

// ---------------------------------------------------------------------------
// One-wave GEMM, 16x64 output tile (one A fragment feeds 4 B fragments per
// k-step: 4x less A traffic, 4 independent WMMAs back-to-back).
// C[rows x O] = act(A[rows x K] @ W[O x K]^T + bias) (+res)
// grid = (rows/16, O/64), block = 32.  A, W f16; bias f32; C f16.
// ---------------------------------------------------------------------------
__global__ __launch_bounds__(32)
void k_gemm(const _Float16* __restrict__ A, const _Float16* __restrict__ W,
            const float* __restrict__ bias, const _Float16* res,
            _Float16* C, int K, int O, int relu) {
  int lane = threadIdx.x;
  int m0 = blockIdx.x * 16;
  int o0 = blockIdx.y * 64;
  v8f acc0 = {}, acc1 = {}, acc2 = {}, acc3 = {};
  for (int kk = 0; kk < K; kk += 32) {
    v16h a = frag_a(A, m0 + (lane & 15), K, kk, lane);
    acc0 = wmma_f16(a, frag_b(W, o0 +  0 + (lane & 15), K, kk, lane), acc0);
    acc1 = wmma_f16(a, frag_b(W, o0 + 16 + (lane & 15), K, kk, lane), acc1);
    acc2 = wmma_f16(a, frag_b(W, o0 + 32 + (lane & 15), K, kk, lane), acc2);
    acc3 = wmma_f16(a, frag_b(W, o0 + 48 + (lane & 15), K, kk, lane), acc3);
  }
  int col = lane & 15;
  int rb = (lane & 16) ? 8 : 0;
#pragma unroll
  for (int ct = 0; ct < 4; ++ct) {
    v8f acc = (ct == 0) ? acc0 : (ct == 1) ? acc1 : (ct == 2) ? acc2 : acc3;
    int oc = o0 + ct * 16 + col;
#pragma unroll
    for (int r = 0; r < 8; ++r) {
      int row = m0 + rb + r;
      float v = acc[r] + bias[oc];
      if (res) v += (float)res[(size_t)row * O + oc];
      if (relu) v = v > 0.f ? v : 0.f;
      C[(size_t)row * O + oc] = (_Float16)v;
    }
  }
}

// ---------------------------------------------------------------------------
// pcat[(b,n), 256] = [time_e | day_e | iep | node_emb]   (f16)
// grid = B*N, block = 64
// ---------------------------------------------------------------------------
__global__ void k_build_pcat(const float* __restrict__ x,
                             const int* __restrict__ time_id,
                             const int* __restrict__ day_id,
                             const float* __restrict__ time_emb,
                             const float* __restrict__ day_emb,
                             const float* __restrict__ node_emb,
                             const float* __restrict__ c1_w,
                             const float* __restrict__ c1_b,
                             _Float16* __restrict__ pcat) {
  int bn = blockIdx.x;
  int b = bn / NN, n = bn % NN;
  int h = threadIdx.x;
  int ti = time_id[(b * TTD + TTD - 1) * NN + n];
  int di = day_id[(b * TTD + TTD - 1) * NN + n];
  float iep = c1_b[h];
#pragma unroll
  for (int t = 0; t < TTD; ++t)
    iep += c1_w[h * TTD + t] * x[((size_t)b * TTD + t) * NN + n];
  _Float16* o = pcat + (size_t)bn * 256;
  o[h]       = (_Float16)time_emb[(size_t)ti * HH + h];
  o[64 + h]  = (_Float16)day_emb[(size_t)di * HH + h];
  o[128 + h] = (_Float16)iep;
  o[192 + h] = (_Float16)node_emb[(size_t)n * HH + h];
}

// ---------------------------------------------------------------------------
// rcat[(b,t,n), 320] = [time_e | day_e | ier | node_emb | pos_emb]  (f16)
// grid = B*T*N, block = 320
// ---------------------------------------------------------------------------
__global__ void k_build_rcat(const float* __restrict__ x,
                             const int* __restrict__ time_id,
                             const int* __restrict__ day_id,
                             const float* __restrict__ time_emb,
                             const float* __restrict__ day_emb,
                             const float* __restrict__ node_emb,
                             const float* __restrict__ pos_emb,
                             const float* __restrict__ dense_w,
                             const float* __restrict__ dense_b,
                             _Float16* __restrict__ rcat) {
  int idx = blockIdx.x;                 // b*T*N + t*N + n
  int n = idx % NN;
  int bt = idx / NN;
  int t = bt % TTD, b = bt / TTD;
  int j = threadIdx.x;                  // 0..319
  int sec = j >> 6, h = j & 63;
  float val;
  if (sec == 0) {
    int ti = time_id[(b * TTD + TTD - 1) * NN + n];
    val = time_emb[(size_t)ti * HH + h];
  } else if (sec == 1) {
    int di = day_id[(b * TTD + TTD - 1) * NN + n];
    val = day_emb[(size_t)di * HH + h];
  } else if (sec == 2) {
    val = dense_b[t * HH + h];
#pragma unroll
    for (int tt = 0; tt < TTD; ++tt)
      val += dense_w[((size_t)t * HH + h) * TTD + tt] *
             x[((size_t)b * TTD + tt) * NN + n];
  } else if (sec == 3) {
    val = node_emb[(size_t)n * HH + h];
  } else {
    val = pos_emb[(size_t)t * HH + h];
  }
  rcat[(size_t)idx * 320 + j] = (_Float16)val;
}

// ---------------------------------------------------------------------------
// memory-module stats: m = softmax(m1, -1), ent[r] = sum m*logm.  1 block x 64.
// ---------------------------------------------------------------------------
__global__ void k_mem_stats(const float* __restrict__ m1,
                            float* __restrict__ m, float* __restrict__ ent) {
  __shared__ float red[64];
  int h = threadIdx.x;
  for (int r = 0; r < MMEM; ++r) {
    float x = m1[r * HH + h];
    red[h] = x; __syncthreads();
    for (int s = 32; s > 0; s >>= 1) {
      if (h < s) red[h] = fmaxf(red[h], red[h + s]);
      __syncthreads();
    }
    float mx = red[0]; __syncthreads();
    float e = __expf(x - mx);
    red[h] = e; __syncthreads();
    for (int s = 32; s > 0; s >>= 1) {
      if (h < s) red[h] += red[h + s];
      __syncthreads();
    }
    float sum = red[0]; __syncthreads();
    float mv = e / sum;
    float lm = x - mx - __logf(sum);
    m[r * HH + h] = mv;
    red[h] = mv * lm; __syncthreads();
    for (int s = 32; s > 0; s >>= 1) {
      if (h < s) red[h] += red[h + s];
      __syncthreads();
    }
    if (h == 0) ent[r] = red[0];
    __syncthreads();
  }
}

// ---------------------------------------------------------------------------
// memory-module assignment: per-row log-softmax, KL argmin, build cat=[emb|g].
// grid = rows, block = 32 (one wave per row; each lane owns h and h+32).
// ---------------------------------------------------------------------------
__global__ __launch_bounds__(32)
void k_mem_assign(const _Float16* __restrict__ embv,
                  const float* __restrict__ m, const float* __restrict__ ent,
                  const float* __restrict__ m1,
                  _Float16* __restrict__ cat, int residual) {
  int row = blockIdx.x;
  int lane = threadIdx.x;
  const _Float16* e = embv + (size_t)row * HH;
  float e0 = (float)e[lane], e1 = (float)e[lane + 32];
  float mx = fmaxf(e0, e1);
  for (int s = 16; s > 0; s >>= 1) mx = fmaxf(mx, __shfl_xor(mx, s, 32));
  float s0 = __expf(e0 - mx), s1 = __expf(e1 - mx);
  float sum = s0 + s1;
  for (int s = 16; s > 0; s >>= 1) sum += __shfl_xor(sum, s, 32);
  float lse = mx + __logf(sum);
  float l0 = e0 - lse, l1 = e1 - lse;
  float best = 3.0e38f; int bi = 0;
  for (int r = 0; r < MMEM; ++r) {
    float d = l0 * m[r * HH + lane] + l1 * m[r * HH + lane + 32];
    for (int s = 16; s > 0; s >>= 1) d += __shfl_xor(d, s, 32);
    float kl = ent[r] - d;
    if (kl < best) { best = kl; bi = r; }
  }
  float g0 = m1[bi * HH + lane], g1 = m1[bi * HH + lane + 32];
  if (residual) { g0 = e0 - g0; g1 = e1 - g1; }
  _Float16* c = cat + (size_t)row * 128;
  c[lane]      = (_Float16)e0;  c[lane + 32] = (_Float16)e1;
  c[64 + lane] = (_Float16)g0;  c[96 + lane] = (_Float16)g1;
}

// ---------------------------------------------------------------------------
// Fused attention: for (b, head d, 16-row n-tile): loop t, S = Q K^T (WMMA),
// scale by hardsig(stru)/8, softmax over m (LDS), O += P V (WMMA), sum over t.
// q/k/v are f16 [B*T*N, 256] with per-head element (n,h) at [.. , h*4+d].
// grid = (20, NHEAD, B), block = 32.  N padded 307 -> 320 with -inf / zeros.
// Next K/V block is prefetched (global_prefetch_b8) while WMMAs run.
// ---------------------------------------------------------------------------
__global__ __launch_bounds__(32)
void k_attention(const _Float16* __restrict__ q, const _Float16* __restrict__ k,
                 const _Float16* __restrict__ v, const float* __restrict__ stru,
                 _Float16* __restrict__ att) {
  __shared__ _Float16 Qs[16 * 64];
  __shared__ _Float16 KVs[64 * 64];   // K block [m][h]; reused as V^T [h][m]
  __shared__ float    Ss[16 * 320];
  __shared__ _Float16 Ps[16 * 320];
  int lane = threadIdx.x;
  int n0 = blockIdx.x * 16;
  int d = blockIdx.y;
  int b = blockIdx.z;
  v8f acc0 = {}, acc1 = {}, acc2 = {}, acc3 = {};

  for (int t = 0; t < TTD; ++t) {
    const size_t base = ((size_t)(b * TTD + t)) * NN * 256;
    // stage Q tile [16 x 64]
    for (int i = lane; i < 16 * 64; i += 32) {
      int rr = i >> 6, h = i & 63;
      int n = n0 + rr;
      Qs[i] = (n < NN) ? q[base + (size_t)n * 256 + h * 4 + d] : (_Float16)0.f;
    }
    __syncthreads();
    // ----- scores -----
    for (int mb = 0; mb < 5; ++mb) {
      for (int i = lane; i < 64 * 64; i += 32) {
        int mm = i >> 6, h = i & 63;
        int mg = mb * 64 + mm;
        KVs[i] = (mg < NN) ? k[base + (size_t)mg * 256 + h * 4 + d] : (_Float16)0.f;
      }
      // prefetch next K block (2 rows per lane) while this block computes
      if (mb < 4) {
        int mg = (mb + 1) * 64 + lane * 2;
        if (mg < NN)     __builtin_prefetch(&k[base + (size_t)mg * 256 + d], 0, 1);
        if (mg + 1 < NN) __builtin_prefetch(&k[base + (size_t)(mg + 1) * 256 + d], 0, 1);
      }
      __syncthreads();
      for (int ct = 0; ct < 4; ++ct) {
        v8f s = {};
        for (int kk = 0; kk < 64; kk += 32) {
          v16h a  = frag_a(Qs, lane & 15, 64, kk, lane);
          v16h bf = frag_b(KVs, ct * 16 + (lane & 15), 64, kk, lane);
          s = wmma_f16(a, bf, s);
        }
        int col = lane & 15;
        int rb = (lane & 16) ? 8 : 0;
        int mg0 = mb * 64 + ct * 16;
#pragma unroll
        for (int r = 0; r < 8; ++r) {
          int rowl = rb + r;
          int n = n0 + rowl;
          int mg = mg0 + col;
          float val;
          if (mg >= NN) {
            val = -1.0e30f;
          } else {
            int nc = (n < NN) ? n : (NN - 1);
            float hs = stru[(size_t)nc * NN + mg] * (1.f / 6.f) + 0.5f;
            hs = fminf(fmaxf(hs, 0.f), 1.f);
            val = s[r] * 0.125f * hs;
          }
          Ss[rowl * 320 + mg] = val;
        }
      }
      __syncthreads();
    }
    // prefetch first V block before softmax
    {
      int mg = lane * 2;
      if (mg < NN)     __builtin_prefetch(&v[base + (size_t)mg * 256 + d], 0, 1);
      if (mg + 1 < NN) __builtin_prefetch(&v[base + (size_t)(mg + 1) * 256 + d], 0, 1);
    }
    // ----- softmax rows -> Ps (f16) -----
    for (int r = 0; r < 16; ++r) {
      float mx = -3.0e38f;
      for (int c = lane; c < 320; c += 32) mx = fmaxf(mx, Ss[r * 320 + c]);
      for (int s = 16; s > 0; s >>= 1) mx = fmaxf(mx, __shfl_xor(mx, s, 32));
      float sum = 0.f;
      for (int c = lane; c < 320; c += 32) {
        float ev = __expf(Ss[r * 320 + c] - mx);
        Ss[r * 320 + c] = ev;
        sum += ev;
      }
      for (int s = 16; s > 0; s >>= 1) sum += __shfl_xor(sum, s, 32);
      float inv = 1.f / sum;
      for (int c = lane; c < 320; c += 32)
        Ps[r * 320 + c] = (_Float16)(Ss[r * 320 + c] * inv);
    }
    __syncthreads();
    // ----- O += P V -----
    for (int mb = 0; mb < 5; ++mb) {
      for (int i = lane; i < 64 * 64; i += 32) {   // stage V transposed [h][m]
        int mm = i >> 6, h = i & 63;
        int mg = mb * 64 + mm;
        KVs[h * 64 + mm] =
            (mg < NN) ? v[base + (size_t)mg * 256 + h * 4 + d] : (_Float16)0.f;
      }
      if (mb < 4) {
        int mg = (mb + 1) * 64 + lane * 2;
        if (mg < NN)     __builtin_prefetch(&v[base + (size_t)mg * 256 + d], 0, 1);
        if (mg + 1 < NN) __builtin_prefetch(&v[base + (size_t)(mg + 1) * 256 + d], 0, 1);
      }
      __syncthreads();
      for (int kk = 0; kk < 64; kk += 32) {
        v16h a = frag_a(Ps, lane & 15, 320, mb * 64 + kk, lane);
        acc0 = wmma_f16(a, frag_b(KVs,  0 + (lane & 15), 64, kk, lane), acc0);
        acc1 = wmma_f16(a, frag_b(KVs, 16 + (lane & 15), 64, kk, lane), acc1);
        acc2 = wmma_f16(a, frag_b(KVs, 32 + (lane & 15), 64, kk, lane), acc2);
        acc3 = wmma_f16(a, frag_b(KVs, 48 + (lane & 15), 64, kk, lane), acc3);
      }
      __syncthreads();
    }
  }
  // store att[(b,n), d*64 + h]  (layout matches transpose(0,2,1,3).reshape)
  int col = lane & 15;
  int rb = (lane & 16) ? 8 : 0;
#pragma unroll
  for (int r = 0; r < 8; ++r) {
    int n = n0 + rb + r;
    if (n < NN) {
      size_t o = ((size_t)b * NN + n) * 256 + d * 64;
      att[o + 0  + col] = (_Float16)acc0[r];
      att[o + 16 + col] = (_Float16)acc1[r];
      att[o + 32 + col] = (_Float16)acc2[r];
      att[o + 48 + col] = (_Float16)acc3[r];
    }
  }
}

// ---------------------------------------------------------------------------
// prediction head: out[b,o,n] = pred_b[o] + pred_w[o,:] . (p + rr)[b,n,:]
// ---------------------------------------------------------------------------
__global__ void k_pred(const _Float16* __restrict__ p,
                       const _Float16* __restrict__ rr,
                       const float* __restrict__ pw,
                       const float* __restrict__ pb,
                       float* __restrict__ out) {
  int idx = blockIdx.x * blockDim.x + threadIdx.x;
  if (idx >= BB * NOUT * NN) return;
  int n = idx % NN;
  int bo = idx / NN;
  int o = bo % NOUT, b = bo / NOUT;
  const _Float16* pp = p + ((size_t)b * NN + n) * HH;
  const _Float16* rp = rr + ((size_t)b * NN + n) * HH;
  float s = pb[o];
#pragma unroll
  for (int h = 0; h < HH; ++h)
    s += pw[o * HH + h] * ((float)pp[h] + (float)rp[h]);
  out[idx] = s;
}

// ---------------------------------------------------------------------------
// host: full pipeline on `stream`
// ---------------------------------------------------------------------------
extern "C" void kernel_launch(void* const* d_in, const int* in_sizes, int n_in,
                              void* d_out, int out_size, void* d_ws, size_t ws_size,
                              hipStream_t stream) {
  (void)in_sizes; (void)n_in; (void)out_size; (void)ws_size;
  const float* xdata   = (const float*)d_in[IN_XDATA];
  const int*   time_id = (const int*)d_in[IN_TIME];
  const int*   day_id  = (const int*)d_in[IN_DAY];
  auto F = [&](int i) { return (const float*)d_in[i]; };

  // ---- workspace bump allocator ----
  char* wsb = (char*)d_ws;
  size_t off = 0;
  auto a16 = [&](size_t n) {
    _Float16* p = (_Float16*)(wsb + off);
    off += (n * sizeof(_Float16) + 255) & ~(size_t)255;
    return p;
  };
  auto a32 = [&](size_t n) {
    float* p = (float*)(wsb + off);
    off += (n * sizeof(float) + 255) & ~(size_t)255;
    return p;
  };

  // f16 weights
  _Float16* w_l1  = a16(64 * 320);
  _Float16* w_ec2 = a16(64 * 256);
  _Float16* w_mc1 = a16(64 * 64);   _Float16* w_mc2 = a16(64 * 128);
  _Float16* w_gmc1 = a16(64 * 64);  _Float16* w_gmc2 = a16(64 * 128);
  _Float16* w_grc1 = a16(64 * 64);  _Float16* w_grc2 = a16(64 * 128);
  _Float16* w_q = a16(256 * 64); _Float16* w_k = a16(256 * 64); _Float16* w_v = a16(256 * 64);
  _Float16* w_d1 = a16(64 * 256);
  _Float16* w_pc1 = a16(NPB * 64 * 64); _Float16* w_pc2 = a16(NPB * 64 * 64);
  _Float16* w_rc1 = a16(NPB * 64 * 64); _Float16* w_rc2 = a16(NPB * 64 * 64);
  // activations
  _Float16* pcat   = a16((size_t)ROWS_BN * 256);
  _Float16* rcat   = a16((size_t)ROWS_BTN * 320);
  _Float16* rbuf   = a16((size_t)ROWS_BTN * 64);
  _Float16* membuf = a16((size_t)ROWS_BTN * 64);
  _Float16* rmembuf = a16((size_t)ROWS_BTN * 64);
  _Float16* embtmp = a16((size_t)ROWS_BTN * 64);
  _Float16* cattmp = a16((size_t)ROWS_BTN * 128);
  _Float16* qbuf = a16((size_t)ROWS_BTN * 256);
  _Float16* kbuf = a16((size_t)ROWS_BTN * 256);
  _Float16* vbuf = a16((size_t)ROWS_BTN * 256);
  _Float16* attbuf = a16((size_t)ROWS_BN * 256);
  _Float16* p0 = a16((size_t)ROWS_BN * 64);
  _Float16* p1 = a16((size_t)ROWS_BN * 64);
  _Float16* ptmp = a16((size_t)ROWS_BN * 64);
  _Float16* rr0 = a16((size_t)ROWS_BN * 64);
  _Float16* rrtmp = a16((size_t)ROWS_BN * 64);
  // f32 memory-module stats (3 instances)
  float* ms_m = a32(MMEM * HH); float* ms_e = a32(MMEM);
  float* gm_m = a32(MMEM * HH); float* gm_e = a32(MMEM);
  float* gr_m = a32(MMEM * HH); float* gr_e = a32(MMEM);

  auto cvt = [&](int idx, _Float16* dst, int n) {
    k_f32_to_f16<<<(n + 255) / 256, 256, 0, stream>>>(F(idx), dst, n);
  };
  cvt(E_L1W, w_l1, 64 * 320);   cvt(E_C2W, w_ec2, 64 * 256);
  cvt(M_C1W, w_mc1, 64 * 64);   cvt(M_C2W, w_mc2, 64 * 128);
  cvt(GM_C1W, w_gmc1, 64 * 64); cvt(GM_C2W, w_gmc2, 64 * 128);
  cvt(GR_C1W, w_grc1, 64 * 64); cvt(GR_C2W, w_grc2, 64 * 128);
  cvt(G_QW, w_q, 256 * 64);     cvt(G_KW, w_k, 256 * 64);
  cvt(G_VW, w_v, 256 * 64);     cvt(G_D1W, w_d1, 64 * 256);
  cvt(P_C1W, w_pc1, NPB * 64 * 64); cvt(P_C2W, w_pc2, NPB * 64 * 64);
  cvt(R_C1W, w_rc1, NPB * 64 * 64); cvt(R_C2W, w_rc2, NPB * 64 * 64);

  // ---- embeddings ----
  k_build_pcat<<<ROWS_BN, 64, 0, stream>>>(xdata, time_id, day_id, F(E_TIMEEMB),
                                           F(E_DAYEMB), F(E_NODE), F(E_C1W),
                                           F(E_C1B), pcat);
  k_gemm<<<dim3(ROWS_BN / 16, 1), 32, 0, stream>>>(pcat, w_ec2, F(E_C2B),
                                                   nullptr, p0, 256, 64, 0);
  k_build_rcat<<<ROWS_BTN, 320, 0, stream>>>(xdata, time_id, day_id,
                                             F(E_TIMEEMB), F(E_DAYEMB),
                                             F(E_NODE), F(E_POS), F(E_DENSEW),
                                             F(E_DENSEB), rcat);
  k_gemm<<<dim3(ROWS_BTN / 16, 1), 32, 0, stream>>>(rcat, w_l1, F(E_L1B),
                                                    nullptr, rbuf, 320, 64, 0);

  // ---- memory modules ----
  auto memory_mod = [&](const _Float16* inp, int rows, const _Float16* wc1,
                        int bc1, const _Float16* wc2, int bc2, int m1i,
                        float* ms, float* me, _Float16* outp, int residual) {
    k_mem_stats<<<1, 64, 0, stream>>>(F(m1i), ms, me);
    k_gemm<<<dim3(rows / 16, 1), 32, 0, stream>>>(inp, wc1, F(bc1), nullptr,
                                                  embtmp, 64, 64, 1);
    k_mem_assign<<<rows, 32, 0, stream>>>(embtmp, ms, me, F(m1i), cattmp,
                                          residual);
    k_gemm<<<dim3(rows / 16, 1), 32, 0, stream>>>(cattmp, wc2, F(bc2), nullptr,
                                                  outp, 128, 64, 0);
  };
  memory_mod(p0, ROWS_BN, w_mc1, M_C1B, w_mc2, M_C2B, M_M1, ms_m, ms_e, p1, 0);

  // period residual blocks on p1
  for (int i = 0; i < NPB; ++i) {
    k_gemm<<<dim3(ROWS_BN / 16, 1), 32, 0, stream>>>(
        p1, w_pc1 + (size_t)i * 4096, F(P_C1B) + i * 64, nullptr, ptmp, 64, 64, 1);
    k_gemm<<<dim3(ROWS_BN / 16, 1), 32, 0, stream>>>(
        ptmp, w_pc2 + (size_t)i * 4096, F(P_C2B) + i * 64, p1, p1, 64, 64, 0);
  }

  // ---- GMA ----
  memory_mod(rbuf, ROWS_BTN, w_gmc1, GM_C1B, w_gmc2, GM_C2B, GM_M1, gm_m, gm_e,
             membuf, 0);
  memory_mod(rbuf, ROWS_BTN, w_grc1, GR_C1B, w_grc2, GR_C2B, GR_M1, gr_m, gr_e,
             rmembuf, 1);
  k_gemm<<<dim3(ROWS_BTN / 16, 4), 32, 0, stream>>>(membuf, w_q, F(G_QB),
                                                    nullptr, qbuf, 64, 256, 1);
  k_gemm<<<dim3(ROWS_BTN / 16, 4), 32, 0, stream>>>(membuf, w_k, F(G_KB),
                                                    nullptr, kbuf, 64, 256, 1);
  k_gemm<<<dim3(ROWS_BTN / 16, 4), 32, 0, stream>>>(rmembuf, w_v, F(G_VB),
                                                    nullptr, vbuf, 64, 256, 1);
  k_attention<<<dim3(20, NHEAD, BB), 32, 0, stream>>>(qbuf, kbuf, vbuf,
                                                      F(G_STRU), attbuf);
  k_gemm<<<dim3(ROWS_BN / 16, 1), 32, 0, stream>>>(attbuf, w_d1, F(G_D1B),
                                                   nullptr, rr0, 256, 64, 1);

  // residual blocks on rr0
  for (int i = 0; i < NPB; ++i) {
    k_gemm<<<dim3(ROWS_BN / 16, 1), 32, 0, stream>>>(
        rr0, w_rc1 + (size_t)i * 4096, F(R_C1B) + i * 64, nullptr, rrtmp, 64, 64, 1);
    k_gemm<<<dim3(ROWS_BN / 16, 1), 32, 0, stream>>>(
        rrtmp, w_rc2 + (size_t)i * 4096, F(R_C2B) + i * 64, rr0, rr0, 64, 64, 0);
  }

  // ---- prediction ----
  k_pred<<<(BB * NOUT * NN + 255) / 256, 256, 0, stream>>>(
      p1, rr0, F(PRED_W), F(PRED_B), (float*)d_out);
}
